// SamplesLoss_76287209111909
// MI455X (gfx1250) — compile-verified, matrix operand order checked by reading, not run on hardware
//
#include <hip/hip_runtime.h>
#include <hip/hip_bf16.h>
#include <math.h>

// Problem constants (match reference)
#define NP     8192
#define DD     64
#define EPS_F  0.05f
#define NITERS 20
#define NSPLIT 8          // column splits per 32-row group (2048 waves/pass)
#define LN2F    0.69314718055994530942f
#define INVLN2F 1.44269504088896340736f

typedef __attribute__((ext_vector_type(8)))  __bf16 bf16x8;
typedef __attribute__((ext_vector_type(16))) __bf16 bf16x16;
typedef __attribute__((ext_vector_type(8)))  float  f32x8;

// ---- single-exp2, select-only online LSE (raw v_exp_f32; args <= 0, FTZ is fine) ----

__device__ __forceinline__ void lse_push(float& m, float& s, float v) {
    float d = v - m;                                       // d>0 => new max
    float w = __builtin_amdgcn_exp2f(-__builtin_fabsf(d)); // neg+abs are src modifiers
    bool gt = d > 0.f;
    s = fmaf(s, gt ? w : 1.0f, gt ? 1.0f : w);
    m = gt ? v : m;                                        // cndmask, no maxnum canonicalize
}

__device__ __forceinline__ void lse_merge(float& m, float& s, float mo, float so) {
    float d = mo - m;
    float w = __builtin_amdgcn_exp2f(-__builtin_fabsf(d));
    bool gt = d > 0.f;
    s = gt ? fmaf(s, w, so) : fmaf(so, w, s);
    m = gt ? mo : m;
}

// ---------------- setup kernels ----------------

__global__ void sums_kernel(const float* __restrict__ a, const float* __restrict__ b,
                            float* __restrict__ scal) {
    __shared__ float red[1024];
    const float* src = (blockIdx.x == 0) ? a : b;
    float acc = 0.f;
    for (int i = threadIdx.x; i < NP; i += 1024) acc += src[i];
    red[threadIdx.x] = acc;
    __syncthreads();
    for (int off = 512; off > 0; off >>= 1) {
        if (threadIdx.x < off) red[threadIdx.x] += red[threadIdx.x + off];
        __syncthreads();
    }
    if (threadIdx.x == 0) {
        scal[blockIdx.x * 2 + 0] = red[0];
        scal[blockIdx.x * 2 + 1] = logf(red[0]);
    }
}

__global__ void prep_kernel(const float* __restrict__ w, const float* __restrict__ pts,
                            const float* __restrict__ logSumPtr,
                            float* __restrict__ logw, float* __restrict__ p2,
                            __bf16* __restrict__ p_hi, __bf16* __restrict__ p_lo) {
    int i = blockIdx.x * blockDim.x + threadIdx.x;
    if (i >= NP) return;
    logw[i] = logf(w[i]) - *logSumPtr;
    float ss = 0.f;
#pragma unroll
    for (int d = 0; d < DD; ++d) {
        float v = pts[i * DD + d];
        ss += v * v;
        __bf16 h = (__bf16)v;
        p_hi[i * DD + d] = h;
        p_lo[i * DD + d] = (__bf16)(v - (float)h);
    }
    p2[i] = ss;
}

__global__ void zero2_kernel(float* __restrict__ f, float* __restrict__ g) {
    int i = blockIdx.x * blockDim.x + threadIdx.x;
    if (i < NP) { f[i] = 0.f; g[i] = 0.f; }
}

// u2[j] = (logw_j + (pot_j - col2_j)/eps) / ln2   (log2 units)
__global__ void ucol_kernel(const float* __restrict__ logw, const float* __restrict__ pot,
                            const float* __restrict__ col2, float* __restrict__ u2) {
    int j = blockIdx.x * blockDim.x + threadIdx.x;
    if (j < NP) u2[j] = (logw[j] + (pot[j] - col2[j]) * (1.0f / EPS_F)) * INVLN2F;
}

// ---------------- fused WMMA soft-min sweep ----------------

__device__ __forceinline__ bf16x16 load_frag(const __bf16* __restrict__ rowPtr, int kbase, int half) {
    // 16-bit A/B fragment: lane-half selects K sub-block; two 16B segments.
    bf16x8 s0 = *(const bf16x8*)(rowPtr + kbase + half * 8);
    bf16x8 s1 = *(const bf16x8*)(rowPtr + kbase + 16 + half * 8);
    return __builtin_shufflevector(s0, s1, 0, 1, 2, 3, 4, 5, 6, 7,
                                   8, 9, 10, 11, 12, 13, 14, 15);
}

#define WMMA_BF16(A, B, C) \
    __builtin_amdgcn_wmma_f32_16x16x32_bf16(false, (A), false, (B), (short)0, (C), false, false)

// One wave owns a 32-row block (two 16-row A tiles) x NP/NSPLIT columns.
// Per column tile: 12 bf16 WMMAs in 2 interleaved accumulation chains
// (one per row tile; D->C accumulation needs no hazard nops), then
// single-exp2 online log2-sum-exp. B fragments are software-pipelined one
// tile ahead (unroll-2 ping-pong); tile count is a runtime argument so the
// loop body stays resident in I-cache (no full unroll).
__launch_bounds__(256)
__global__ void softmin_pass_kernel(const __bf16* __restrict__ q_hi, const __bf16* __restrict__ q_lo,
                                    const __bf16* __restrict__ k_hi, const __bf16* __restrict__ k_lo,
                                    const float* __restrict__ u2, float2* __restrict__ part,
                                    int tilesPerSplit) {
    const int lane  = threadIdx.x & 31;
    const int wave  = threadIdx.x >> 5;
    const int gwave = blockIdx.x * 8 + wave;
    const int rowGrp = gwave / NSPLIT;
    const int split  = gwave % NSPLIT;
    const int rowBase = rowGrp * 32;
    const int r    = lane & 15;
    const int half = lane >> 4;

    const __bf16* q0h = q_hi + (size_t)(rowBase + r) * DD;
    const __bf16* q0l = q_lo + (size_t)(rowBase + r) * DD;
    const __bf16* q1h = q_hi + (size_t)(rowBase + 16 + r) * DD;
    const __bf16* q1l = q_lo + (size_t)(rowBase + 16 + r) * DD;
    // A fragments resident for the whole sweep: 2 row tiles x (hi,lo) x (K0,K1)
    bf16x16 a0H0 = load_frag(q0h, 0, half),  a0H1 = load_frag(q0h, 32, half);
    bf16x16 a0L0 = load_frag(q0l, 0, half),  a0L1 = load_frag(q0l, 32, half);
    bf16x16 a1H0 = load_frag(q1h, 0, half),  a1H1 = load_frag(q1h, 32, half);
    bf16x16 a1L0 = load_frag(q1l, 0, half),  a1L1 = load_frag(q1l, 32, half);

    float m[16], sm[16];
#pragma unroll
    for (int v = 0; v < 16; ++v) { m[v] = -1e30f; sm[v] = 0.f; }

    const float scale2 = 2.0f * (1.0f / EPS_F) * INVLN2F;

    // compute one 32x16 tile: 12 WMMAs + online LSE epilogue
    auto tile_compute = [&](bf16x16 tH0, bf16x16 tH1, bf16x16 tL0, bf16x16 tL1, float tuj) {
        f32x8 cA = {}, cB = {};
        cA = WMMA_BF16(a0L0, tH0, cA);  cB = WMMA_BF16(a1L0, tH0, cB);
        cA = WMMA_BF16(a0H0, tL0, cA);  cB = WMMA_BF16(a1H0, tL0, cB);
        cA = WMMA_BF16(a0H0, tH0, cA);  cB = WMMA_BF16(a1H0, tH0, cB);
        cA = WMMA_BF16(a0L1, tH1, cA);  cB = WMMA_BF16(a1L1, tH1, cB);
        cA = WMMA_BF16(a0H1, tL1, cA);  cB = WMMA_BF16(a1H1, tL1, cB);
        cA = WMMA_BF16(a0H1, tH1, cA);  cB = WMMA_BF16(a1H1, tH1, cB);
#pragma unroll
        for (int v = 0; v < 8; ++v)
            lse_push(m[v], sm[v], fmaf(cA[v], scale2, tuj));
#pragma unroll
        for (int v = 0; v < 8; ++v)
            lse_push(m[8 + v], sm[8 + v], fmaf(cB[v], scale2, tuj));
    };

    // pipeline prologue: load tile 0's B fragments + column constant
    int colBase = split * tilesPerSplit * 16;
    bf16x16 bH0 = load_frag(k_hi + (size_t)(colBase + r) * DD, 0, half);
    bf16x16 bH1 = load_frag(k_hi + (size_t)(colBase + r) * DD, 32, half);
    bf16x16 bL0 = load_frag(k_lo + (size_t)(colBase + r) * DD, 0, half);
    bf16x16 bL1 = load_frag(k_lo + (size_t)(colBase + r) * DD, 32, half);
    float uj = u2[colBase + r];

#pragma unroll 2
    for (int t = 0; t < tilesPerSplit - 1; ++t) {
        // unconditional prefetch of tile t+1 (always in-bounds; affine address)
        const int nextBase = colBase + 16;
        const __bf16* nkh = k_hi + (size_t)(nextBase + r) * DD;
        const __bf16* nkl = k_lo + (size_t)(nextBase + r) * DD;
        bf16x16 nH0 = load_frag(nkh, 0, half), nH1 = load_frag(nkh, 32, half);
        bf16x16 nL0 = load_frag(nkl, 0, half), nL1 = load_frag(nkl, 32, half);
        float un = u2[nextBase + r];

        tile_compute(bH0, bH1, bL0, bL1, uj);

        // rotate pipeline registers (dissolved by unroll-2 renaming)
        bH0 = nH0; bH1 = nH1; bL0 = nL0; bL1 = nL1;
        uj = un;
        colBase = nextBase;
    }
    // last tile: no prefetch
    tile_compute(bH0, bH1, bL0, bL1, uj);

    // merge across the 16 lanes that share each row (C/D layout: lane = column)
#pragma unroll
    for (int v = 0; v < 16; ++v) {
        float mv = m[v], sv = sm[v];
#pragma unroll
        for (int off = 1; off < 16; off <<= 1) {
            float mo = __shfl_xor(mv, off, 32);
            float so = __shfl_xor(sv, off, 32);
            lse_merge(mv, sv, mo, so);
        }
        m[v] = mv; sm[v] = sv;
    }
    if (r == 0) {
#pragma unroll
        for (int v = 0; v < 8; ++v) {
            int row0 = rowBase + v + half * 8;
            part[(size_t)row0 * NSPLIT + split] = make_float2(m[v], sm[v]);
            int row1 = rowBase + 16 + v + half * 8;
            part[(size_t)row1 * NSPLIT + split] = make_float2(m[8 + v], sm[8 + v]);
        }
    }
}

// Merge NSPLIT partial (max, sum) pairs -> potential: pot_i = q2_i - eps*ln2*(m + log2 s)
__global__ void combine_kernel(const float2* __restrict__ part, const float* __restrict__ q2,
                               float* __restrict__ pot) {
    int i = blockIdx.x * blockDim.x + threadIdx.x;
    if (i >= NP) return;
    float m = -1e30f, s = 0.f;
#pragma unroll
    for (int k = 0; k < NSPLIT; ++k) {
        float2 p = part[(size_t)i * NSPLIT + k];
        lse_merge(m, s, p.x, p.y);
    }
    pot[i] = q2[i] - EPS_F * LN2F * (m + log2f(s));
}

// out = <a/sumA, f_fin> + <b/sumB, g_fin>  (deterministic single-block reduce)
__global__ void final_kernel(const float* __restrict__ a, const float* __restrict__ b,
                             const float* __restrict__ scal,
                             const float* __restrict__ ff, const float* __restrict__ gf,
                             float* __restrict__ out) {
    __shared__ float red[1024];
    const float invA = 1.0f / scal[0];
    const float invB = 1.0f / scal[2];
    float acc = 0.f;
    for (int i = threadIdx.x; i < NP; i += 1024)
        acc += a[i] * invA * ff[i] + b[i] * invB * gf[i];
    red[threadIdx.x] = acc;
    __syncthreads();
    for (int off = 512; off > 0; off >>= 1) {
        if (threadIdx.x < off) red[threadIdx.x] += red[threadIdx.x + off];
        __syncthreads();
    }
    if (threadIdx.x == 0) out[0] = red[0];
}

// ---------------- host orchestration ----------------

extern "C" void kernel_launch(void* const* d_in, const int* in_sizes, int n_in,
                              void* d_out, int out_size, void* d_ws, size_t ws_size,
                              hipStream_t stream) {
    const float* a = (const float*)d_in[0]; // [N]
    const float* x = (const float*)d_in[1]; // [N,D]
    const float* b = (const float*)d_in[2]; // [M]
    const float* y = (const float*)d_in[3]; // [M,D]
    float* out = (float*)d_out;

    // workspace carve-up (~5 MB)
    char* w = (char*)d_ws;
    float*  scal = (float*)w;                    // [0]=sumA [1]=logSumA [2]=sumB [3]=logSumB
    float*  loga = (float*)(w + 256);
    float*  logb = loga + NP;
    float*  x2   = logb + NP;
    float*  y2   = x2 + NP;
    float*  f    = y2 + NP;
    float*  g    = f + NP;
    float*  ffin = g + NP;
    float*  gfin = ffin + NP;
    float*  u2   = gfin + NP;
    float2* part = (float2*)(u2 + NP);           // NP*NSPLIT pairs
    __bf16* xh = (__bf16*)(part + (size_t)NP * NSPLIT);
    __bf16* xl = xh + (size_t)NP * DD;
    __bf16* yh = xl + (size_t)NP * DD;
    __bf16* yl = yh + (size_t)NP * DD;

    const int T = 256;
    const int VB = (NP + T - 1) / T;                  // 32 blocks for N-length vectors
    const int PASS_BLOCKS = ((NP / 32) * NSPLIT) / 8; // 256 blocks, 8 waves each
    const int TILES = (NP / 16) / NSPLIT;             // 64 (runtime arg: prevents full unroll)

    // setup
    sums_kernel<<<2, 1024, 0, stream>>>(a, b, scal);
    prep_kernel<<<VB, T, 0, stream>>>(a, x, scal + 1, loga, x2, xh, xl);
    prep_kernel<<<VB, T, 0, stream>>>(b, y, scal + 3, logb, y2, yh, yl);
    zero2_kernel<<<VB, T, 0, stream>>>(f, g);

    auto rows_pass = [&](float* fOut) { // f = softmin over columns (uses g, logb, y2)
        ucol_kernel<<<VB, T, 0, stream>>>(logb, g, y2, u2);
        softmin_pass_kernel<<<PASS_BLOCKS, T, 0, stream>>>(xh, xl, yh, yl, u2, part, TILES);
        combine_kernel<<<VB, T, 0, stream>>>(part, x2, fOut);
    };
    auto cols_pass = [&](const float* fIn, float* gOut) { // g = softmin over rows
        ucol_kernel<<<VB, T, 0, stream>>>(loga, fIn, x2, u2);
        softmin_pass_kernel<<<PASS_BLOCKS, T, 0, stream>>>(yh, yl, xh, xl, u2, part, TILES);
        combine_kernel<<<VB, T, 0, stream>>>(part, y2, gOut);
    };

    for (int it = 0; it < NITERS; ++it) {
        rows_pass(f);       // f <- softmin_rows(g)
        cols_pass(f, g);    // g <- softmin_cols(f)
    }
    // final symmetric update from (detached) potentials
    rows_pass(ffin);        // f_fin = softmin_rows(g)
    cols_pass(f, gfin);     // g_fin = softmin_cols(f)

    final_kernel<<<1, 1024, 0, stream>>>(a, b, scal, ffin, gfin, out);
}
